// SymbolicOperator_80212809220431
// MI455X (gfx1250) — matrix operand order repaired and checked by reading, not compiled, not run on hardware
//
#include <hip/hip_runtime.h>
#include <hip/hip_bf16.h>
#include <math.h>

// ---------------------------------------------------------------------------
// Sizes from the reference
// ---------------------------------------------------------------------------
#define BATCH   512
#define NWORDS  15
#define STEPS   3
#define MAXLEN  50
#define KEYDIM  128
#define NPTR    3
#define HID     384            // KEYDIM * NPTR
#define G3      1152           // 3 * HID
#define OUTV    256
#define PDIM    200
#define INDIM   456            // PDIM + OUTV
#define KIH     480            // INDIM padded to multiple of 32
#define KHH     384            // HID (already multiple of 32)

#define BM      128            // GEMM block M tile
#define BN      64             // GEMM block N tile (4 WMMA n-subtiles per wave)
#define NBLKN   (G3 / BN)      // 18 n-blocks
#define NBLKM   (BATCH / BM)   // 4 m-blocks
#define BLKS1   (NBLKM * NBLKN) // 72 blocks per GEMM

typedef __attribute__((ext_vector_type(16))) _Float16 v16h;
typedef __attribute__((ext_vector_type(8)))  float    v8f;

struct F16x16 { float4 lo; float4 hi; };   // 32 bytes == v16h

__device__ __forceinline__ float sigmoidf_(float x) { return 1.0f / (1.0f + __expf(-x)); }

// ---------------------------------------------------------------------------
// Init kernels
// ---------------------------------------------------------------------------
__global__ void k_init_keys(float* __restrict__ keys) {
    int i = blockIdx.x * 256 + threadIdx.x;
    if (i >= MAXLEN * KEYDIM) return;
    int l = i >> 7, k = i & 127;
    int i2 = k >> 1;
    float div = __expf(-(float)(2 * i2) * (logf(10000.0f) / (float)KEYDIM));
    float ang = (float)l * div;
    keys[i] = (k & 1) ? cosf(ang) : sinf(ang);
}

__global__ void k_init_hidden(const float* __restrict__ keys,
                              float* __restrict__ hidden,
                              _Float16* __restrict__ H16) {
    int i = blockIdx.x * 256 + threadIdx.x;
    if (i >= BATCH * HID) return;
    int j = i % HID;
    float v = keys[j % KEYDIM];          // keys row 0, tiled 3x
    hidden[i] = v;
    H16[i] = (_Float16)v;
}

__global__ void k_init_sv(float* __restrict__ sv) {
    int i = blockIdx.x * 256 + threadIdx.x;
    if (i >= BATCH * MAXLEN * OUTV) return;
    sv[i] = ((i & (OUTV - 1)) == 0) ? 1.0f : 0.0f;   // one-hot at EOS_IDX=0
}

__global__ void k_cvt_wih(const float* __restrict__ w, _Float16* __restrict__ w16) {
    int i = blockIdx.x * 256 + threadIdx.x;
    if (i >= G3 * KIH) return;
    int n = i / KIH, k = i % KIH;
    w16[i] = (_Float16)((k < INDIM) ? w[n * INDIM + k] : 0.0f);
}

__global__ void k_cvt_whh(const float* __restrict__ w, _Float16* __restrict__ w16) {
    int i = blockIdx.x * 256 + threadIdx.x;
    if (i >= G3 * KHH) return;
    w16[i] = (_Float16)w[i];
}

// ---------------------------------------------------------------------------
// Attention + scratch-value update + X build (one block per batch element)
// ---------------------------------------------------------------------------
__global__ void __launch_bounds__(256)
k_attn_sv(const int* __restrict__ inst,
          const float* __restrict__ gate_emb,
          const float* __restrict__ prog_emb,
          const float* __restrict__ prim_emb,
          const float* __restrict__ keys,
          const float* __restrict__ hidden,
          float* __restrict__ sv,
          _Float16* __restrict__ X16,
          int word) {
    __shared__ float sh[HID];
    __shared__ float slr[MAXLEN], slw[MAXLEN], sar[MAXLEN], saw[MAXLEN];
    __shared__ float srv[OUTV], snv[OUTV];
    __shared__ float sred[2];

    const int t = threadIdx.x;
    const int b = blockIdx.x;

    for (int j = t; j < HID; j += 256) sh[j] = hidden[(size_t)b * HID + j];
    __syncthreads();

    const int idx = inst[word * BATCH + b];
    const float gate = sigmoidf_(gate_emb[idx]);

    // attention logits: read_ptr = hidden[128:256], write_ptr = hidden[256:384]
    if (t < MAXLEN) {
        const float* kr = keys + t * KEYDIM;
        float acc = 0.0f;
        #pragma unroll 8
        for (int k = 0; k < KEYDIM; ++k) acc += sh[KEYDIM + k] * kr[k];
        slr[t] = acc;
    } else if (t >= 64 && t < 64 + MAXLEN) {
        const int l = t - 64;
        const float* kr = keys + l * KEYDIM;
        float acc = 0.0f;
        #pragma unroll 8
        for (int k = 0; k < KEYDIM; ++k) acc += sh[2 * KEYDIM + k] * kr[k];
        slw[l] = acc;
    }
    __syncthreads();

    // softmax over 50 positions (tiny; two lanes do the serial reductions)
    if (t == 0) {
        float m = -1e30f;
        for (int l = 0; l < MAXLEN; ++l) m = fmaxf(m, slr[l]);
        float s = 0.0f;
        for (int l = 0; l < MAXLEN; ++l) { float e = __expf(slr[l] - m); sar[l] = e; s += e; }
        sred[0] = 1.0f / s;
    } else if (t == 32) {
        float m = -1e30f;
        for (int l = 0; l < MAXLEN; ++l) m = fmaxf(m, slw[l]);
        float s = 0.0f;
        for (int l = 0; l < MAXLEN; ++l) { float e = __expf(slw[l] - m); saw[l] = e; s += e; }
        sred[1] = 1.0f / s;
    }
    __syncthreads();
    if (t < MAXLEN) { sar[t] *= sred[0]; saw[t] *= sred[1]; }
    __syncthreads();

    // read_value (thread t == channel v); coalesced across v
    float* svb = sv + (size_t)b * MAXLEN * OUTV;
    {
        float rv = 0.0f;
        #pragma unroll 5
        for (int l = 0; l < MAXLEN; ++l) rv += sar[l] * svb[l * OUTV + t];
        srv[t] = rv;
        snv[t] = gate * prim_emb[idx * OUTV + t] + (1.0f - gate) * rv;
    }
    __syncthreads();

    // scratch-value update: sv = wm*new + (1-wm)*sv
    {
        const float nv = snv[t];
        #pragma unroll 5
        for (int l = 0; l < MAXLEN; ++l) {
            const float wm = saw[l];
            const float o  = svb[l * OUTV + t];
            svb[l * OUTV + t] = wm * nv + (1.0f - wm) * o;
        }
    }

    // x = concat(program, read_value), zero-padded K to 480, as f16 for WMMA
    for (int k = t; k < KIH; k += 256) {
        float xv;
        if (k < PDIM)       xv = prog_emb[idx * PDIM + k];
        else if (k < INDIM) xv = srv[k - PDIM];
        else                xv = 0.0f;
        X16[(size_t)b * KIH + k] = (_Float16)xv;
    }
}

// ---------------------------------------------------------------------------
// Fused WMMA GEMM pair with async global->LDS double-buffered staging.
//   blocks [0,72):   C0[M,N] = A0[M,K0] * W0[N,K0]^T   (GI,  K0 = 480)
//   blocks [72,144): C1[M,N] = A1[M,K1] * W1[N,K1]^T   (GH,  K1 = 384)
// Block tile 128x64; 8 waves, wave w -> rows [w*16, w*16+16), 4 n-subtiles.
// Per K-chunk (32): stage A (128x32 f16, 8KB) + B (64x32 f16, 4KB) into LDS
// with 3 GLOBAL_LOAD_ASYNC_TO_LDS_B128 per thread, double buffered so the
// async copies of chunk c+1 overlap the WMMAs of chunk c.
// A reg layout (16-bit A 16x32): e0..7 -> k=hi*8+e ; e8..15 -> k=16+hi*8+(e-8)
// B reg layout (16-bit B 32x16): e0..7 -> k=hi*8+e ; e8..15 -> k=16+hi*8+(e-8), col = lane&15 (B[k][n]=W[n][k])
// ---------------------------------------------------------------------------
__global__ void __launch_bounds__(256)
k_gemm_pair(const _Float16* __restrict__ A0, const _Float16* __restrict__ W0,
            float* __restrict__ C0, int K0,
            const _Float16* __restrict__ A1, const _Float16* __restrict__ W1,
            float* __restrict__ C1, int K1) {
    __shared__ __align__(64) _Float16 sA[2][BM * 32];   // 8KB per buffer
    __shared__ __align__(64) _Float16 sB[2][BN * 32];   // 4KB per buffer

    int g = blockIdx.x;
    const _Float16* A; const _Float16* W; float* C; int Kp;
    if (g < BLKS1) { A = A0; W = W0; C = C0; Kp = K0; }
    else           { g -= BLKS1; A = A1; W = W1; C = C1; Kp = K1; }

    const int t    = threadIdx.x;
    const int wave = t >> 5, lane = t & 31;
    const int hi   = lane >> 4, mn = lane & 15;
    const int bm0  = (g / NBLKN) * BM;
    const int bn0  = (g % NBLKN) * BN;

    // staging geometry: each thread copies 16B per shot
    const int ar0 = t >> 2;              // A shot0 row (0..63)
    const int ar1 = ar0 + 64;            // A shot1 row (64..127)
    const int ac  = (t & 3) * 8;         // k-subchunk (elements)

    auto stage = [&](int buf, int kk) {
        {
            unsigned d = (unsigned)(uintptr_t)&sA[buf][ar0 * 32 + ac];
            unsigned long long s =
                (unsigned long long)(uintptr_t)(A + (size_t)(bm0 + ar0) * Kp + kk + ac);
            asm volatile("global_load_async_to_lds_b128 %0, %1, off"
                         :: "v"(d), "v"(s) : "memory");
        }
        {
            unsigned d = (unsigned)(uintptr_t)&sA[buf][ar1 * 32 + ac];
            unsigned long long s =
                (unsigned long long)(uintptr_t)(A + (size_t)(bm0 + ar1) * Kp + kk + ac);
            asm volatile("global_load_async_to_lds_b128 %0, %1, off"
                         :: "v"(d), "v"(s) : "memory");
        }
        {
            unsigned d = (unsigned)(uintptr_t)&sB[buf][ar0 * 32 + ac];   // rows 0..63
            unsigned long long s =
                (unsigned long long)(uintptr_t)(W + (size_t)(bn0 + ar0) * Kp + kk + ac);
            asm volatile("global_load_async_to_lds_b128 %0, %1, off"
                         :: "v"(d), "v"(s) : "memory");
        }
    };

    v8f acc[4] = {};
    const int nc = Kp / 32;
    stage(0, 0);

    for (int c = 0; c < nc; ++c) {
        const int buf = c & 1;
        if (c + 1 < nc) {
            stage(buf ^ 1, (c + 1) * 32);
            // previous stage's 3 async ops done once <= 3 remain outstanding
            asm volatile("s_wait_asynccnt 0x3" ::: "memory");
        } else {
            asm volatile("s_wait_asynccnt 0x0" ::: "memory");
        }
        __syncthreads();   // all waves' stages of `buf` complete & visible

        const _Float16* aw = &sA[buf][(wave * 16 + mn) * 32];
        F16x16 ap;
        ap.lo = *(const float4*)(aw + hi * 8);
        ap.hi = *(const float4*)(aw + 16 + hi * 8);
        v16h av = __builtin_bit_cast(v16h, ap);

        #pragma unroll
        for (int nt = 0; nt < 4; ++nt) {
            const _Float16* bw = &sB[buf][(nt * 16 + mn) * 32];
            F16x16 bp;
            bp.lo = *(const float4*)(bw + hi * 8);
            bp.hi = *(const float4*)(bw + 16 + hi * 8);
            v16h bv = __builtin_bit_cast(v16h, bp);
            acc[nt] = __builtin_amdgcn_wmma_f32_16x16x32_f16(
                /*neg_a=*/false, av, /*neg_b=*/false, bv,
                /*c_mod=*/(short)0, acc[nt], /*reuse_a=*/false, /*reuse_b=*/false);
        }
        __syncthreads();   // all waves done reading `buf` before it is restaged
    }

    // D layout: VGPR r -> row +r+8*hi, col = lane&15
    #pragma unroll
    for (int nt = 0; nt < 4; ++nt)
        #pragma unroll
        for (int r = 0; r < 8; ++r)
            C[(size_t)(bm0 + wave * 16 + r + 8 * hi) * G3 + bn0 + nt * 16 + mn] = acc[nt][r];
}

// ---------------------------------------------------------------------------
// GRU pointwise update
// ---------------------------------------------------------------------------
__global__ void k_gru(const float* __restrict__ GI, const float* __restrict__ GH,
                      const float* __restrict__ b_ih, const float* __restrict__ b_hh,
                      float* __restrict__ hidden, _Float16* __restrict__ H16) {
    int i = blockIdx.x * 256 + threadIdx.x;
    if (i >= BATCH * HID) return;
    const int b = i / HID, j = i % HID;
    const size_t base = (size_t)b * G3;
    const float r  = sigmoidf_(GI[base + j]           + b_ih[j] +
                               GH[base + j]           + b_hh[j]);
    const float z  = sigmoidf_(GI[base + HID + j]     + b_ih[HID + j] +
                               GH[base + HID + j]     + b_hh[HID + j]);
    const float hn = GH[base + 2 * HID + j] + b_hh[2 * HID + j];
    const float n  = tanhf(GI[base + 2 * HID + j] + b_ih[2 * HID + j] + r * hn);
    const float h  = (1.0f - z) * n + z * hidden[i];
    hidden[i] = h;
    H16[i] = (_Float16)h;
}

// ---------------------------------------------------------------------------
// Final: actions = log_softmax(sv, axis=v) transposed to (B, V, L)
// ---------------------------------------------------------------------------
__global__ void __launch_bounds__(256)
k_logsoftmax_t(const float* __restrict__ sv, float* __restrict__ out) {
    const int row = blockIdx.x;              // b*50 + l
    const int b = row / MAXLEN, l = row % MAXLEN;
    const int t = threadIdx.x;
    __shared__ float red[256];
    const float x = sv[(size_t)row * OUTV + t];
    red[t] = x; __syncthreads();
    for (int s = 128; s > 0; s >>= 1) { if (t < s) red[t] = fmaxf(red[t], red[t + s]); __syncthreads(); }
    const float m = red[0]; __syncthreads();
    red[t] = __expf(x - m); __syncthreads();
    for (int s = 128; s > 0; s >>= 1) { if (t < s) red[t] += red[t + s]; __syncthreads(); }
    const float lse = m + __logf(red[0]);
    out[(size_t)b * OUTV * MAXLEN + (size_t)t * MAXLEN + l] = x - lse;
}

__global__ void k_copy_ta(const int* __restrict__ ta, float* __restrict__ out) {
    int i = blockIdx.x * 256 + threadIdx.x;   // b*50 + l
    if (i >= BATCH * MAXLEN) return;
    const int b = i / MAXLEN, l = i % MAXLEN;
    out[i] = (float)ta[l * BATCH + b];        // true_actions.T
}

// ---------------------------------------------------------------------------
// Launch
// ---------------------------------------------------------------------------
extern "C" void kernel_launch(void* const* d_in, const int* in_sizes, int n_in,
                              void* d_out, int out_size, void* d_ws, size_t ws_size,
                              hipStream_t stream) {
    const int*   inst     = (const int*)  d_in[0];
    const int*   ta       = (const int*)  d_in[1];
    const float* gate_emb = (const float*)d_in[2];
    const float* prog_emb = (const float*)d_in[3];
    const float* prim_emb = (const float*)d_in[4];
    const float* w_ih     = (const float*)d_in[5];
    const float* w_hh     = (const float*)d_in[6];
    const float* b_ih     = (const float*)d_in[7];
    const float* b_hh     = (const float*)d_in[8];
    float* out = (float*)d_out;

    // workspace carve-out (256B aligned)
    char* ws = (char*)d_ws;
    size_t off = 0;
    auto alloc = [&](size_t bytes) -> char* {
        char* p = ws + off;
        off = (off + bytes + 255) & ~(size_t)255;
        return p;
    };
    float*    keys   = (float*)   alloc((size_t)MAXLEN * KEYDIM * 4);
    float*    hidden = (float*)   alloc((size_t)BATCH * HID * 4);
    float*    sv     = (float*)   alloc((size_t)BATCH * MAXLEN * OUTV * 4);
    _Float16* X16    = (_Float16*)alloc((size_t)BATCH * KIH * 2);
    _Float16* H16    = (_Float16*)alloc((size_t)BATCH * KHH * 2);
    _Float16* Wih16  = (_Float16*)alloc((size_t)G3 * KIH * 2);
    _Float16* Whh16  = (_Float16*)alloc((size_t)G3 * KHH * 2);
    float*    GI     = (float*)   alloc((size_t)BATCH * G3 * 4);
    float*    GH     = (float*)   alloc((size_t)BATCH * G3 * 4);

    k_init_keys  <<<(MAXLEN * KEYDIM + 255) / 256, 256, 0, stream>>>(keys);
    k_init_hidden<<<(BATCH * HID + 255) / 256,     256, 0, stream>>>(keys, hidden, H16);
    k_init_sv    <<<(BATCH * MAXLEN * OUTV + 255) / 256, 256, 0, stream>>>(sv);
    k_cvt_wih    <<<(G3 * KIH + 255) / 256,        256, 0, stream>>>(w_ih, Wih16);
    k_cvt_whh    <<<(G3 * KHH + 255) / 256,        256, 0, stream>>>(w_hh, Whh16);

    for (int w = 0; w < NWORDS; ++w) {
        for (int s = 0; s < STEPS; ++s) {
            k_attn_sv<<<BATCH, 256, 0, stream>>>(inst, gate_emb, prog_emb, prim_emb,
                                                 keys, hidden, sv, X16, w);
            k_gemm_pair<<<2 * BLKS1, 256, 0, stream>>>(X16, Wih16, GI, KIH,
                                                       H16, Whh16, GH, KHH);
            k_gru<<<(BATCH * HID + 255) / 256, 256, 0, stream>>>(GI, GH, b_ih, b_hh,
                                                                 hidden, H16);
        }
    }

    k_logsoftmax_t<<<BATCH * MAXLEN, 256, 0, stream>>>(sv, out);
    k_copy_ta<<<(BATCH * MAXLEN + 255) / 256, 256, 0, stream>>>(
        ta, out + (size_t)BATCH * OUTV * MAXLEN);
}
